// Self_Attention_Bidaf_81965155877541
// MI455X (gfx1250) — matrix operand order repaired and checked by reading, not compiled, not run on hardware
//
#include <hip/hip_runtime.h>

// ---------------------------------------------------------------------------
// MI455X / gfx1250, wave32.  bf16 WMMA (v_wmma_f32_16x16x32_bf16) for all
// GEMM-like work; fp32 accumulation and fp32 gate math.  GEMM A-tiles are
// staged into LDS with GLOBAL_LOAD_ASYNC_TO_LDS (ASYNCcnt) double buffering.
// ---------------------------------------------------------------------------

typedef __attribute__((ext_vector_type(16))) __bf16 v16bf;
typedef __attribute__((ext_vector_type(8)))  float  v8f;

union Frag16 {           // one WMMA 16-lane bf16 operand fragment (8 VGPRs)
  v16bf v;
  uint4 q[2];
};

__device__ inline unsigned short f2bf(float f) {
  unsigned int u = __float_as_uint(f);
  unsigned int r = u + 0x7FFFu + ((u >> 16) & 1u);   // round-to-nearest-even
  return (unsigned short)(r >> 16);
}

__device__ inline float sigmoidf_(float x) { return 1.f / (1.f + __expf(-x)); }

// Generic (flat) LDS pointer -> LDS byte offset: low 32 bits per aperture rule.
__device__ inline unsigned lds_off(const void* p) {
  return (unsigned)(unsigned long long)p;
}

// ---------------------------------------------------------------------------
// fp32 -> bf16 (ushort storage) conversion
// ---------------------------------------------------------------------------
__global__ void cvt_f32_bf16(const float* __restrict__ in,
                             unsigned short* __restrict__ out, size_t n) {
  size_t i = (size_t)blockIdx.x * blockDim.x + threadIdx.x;
  if (i < n) out[i] = f2bf(in[i]);
}

// ---------------------------------------------------------------------------
// GEMM: out[r,c] = bias[c] + sum_k A[r,k] * W[c,k]
//   A : (R, K)  bf16 row-major      (activations)
//   W : (NC,K)  bf16 row-major      (weights, einsum 'rd,cd->rc')
//   out: (R,NC) fp32
// Block = 256 threads = 8 waves; block tile 32(M) x 256(N); wave tile 16x64.
// grid = (R/32, NC/256).  K % 32 == 0.
// A block-tile (32 rows x 32 K bf16 = 2 KB) is async-staged into LDS, double
// buffered; B streams straight from L2 (weights are L2-resident).
// ---------------------------------------------------------------------------
__global__ __launch_bounds__(256)
void gemm_bf16_nt(const unsigned short* __restrict__ A,
                  const unsigned short* __restrict__ W,
                  const float* __restrict__ bias,
                  float* __restrict__ out, int K, int NC) {
  __shared__ unsigned short Ablk[2][32 * 32];   // [buf][row*32 + k]

  const int wave  = threadIdx.x >> 5;
  const int lane  = threadIdx.x & 31;
  const int lrow  = lane & 15;
  const bool hi   = lane >= 16;
  const int rBase0 = blockIdx.x * 32;
  const int mHalf  = (wave & 1) * 16;           // which 16-row half of tile
  const int cBase  = blockIdx.y * 256 + (wave >> 1) * 64;

  const v8f vzero = {0.f,0.f,0.f,0.f,0.f,0.f,0.f,0.f};
  v8f acc[4];
  #pragma unroll
  for (int nt = 0; nt < 4; ++nt) acc[nt] = vzero;

  const int khalfA = hi ? 8 : 0;
  const int khalfB = hi ? 8 : 0;
  const unsigned ldsBase = lds_off(&Ablk[0][0]);

  // async-stage one 32x32 bf16 A tile (threads 0..127 issue one b128 each)
  auto stageA = [&](int buf, int k0) {
    const int tid = threadIdx.x;
    if (tid < 128) {
      const int row = tid >> 2;                 // 0..31
      const int seg = (tid & 3) * 8;            // ushort offset, 16B segments
      const unsigned short* g = A + (size_t)(rBase0 + row) * K + k0 + seg;
      const unsigned l = ldsBase + (unsigned)buf * (32 * 32 * 2) +
                         (unsigned)(row * 32 + seg) * 2;
      asm volatile("global_load_async_to_lds_b128 %0, %1, off"
                   :: "v"(l), "v"(g) : "memory");
    }
  };

  stageA(0, 0);
  asm volatile("s_wait_asynccnt 0x0" ::: "memory");
  __syncthreads();

  int buf = 0;
  for (int k0 = 0; k0 < K; k0 += 32) {
    if (k0 + 32 < K) stageA(buf ^ 1, k0 + 32);  // overlap with compute

    // A fragment: 16x32 bf16 from LDS, lane holds row (lane&15); K-halves
    // split across lane groups per ISA layout.
    Frag16 a;
    const unsigned short* ar = &Ablk[buf][(mHalf + lrow) * 32 + khalfA];
    a.q[0] = *(const uint4*)(ar);
    a.q[1] = *(const uint4*)(ar + 16);

    #pragma unroll
    for (int nt = 0; nt < 4; ++nt) {
      // B fragment: 32x16 bf16, lane = output column, K split like A fragment
      const unsigned short* wRow =
          W + (size_t)(cBase + nt * 16 + lrow) * K + k0 + khalfB;
      __builtin_prefetch(wRow + 64, 0, 1);      // global_prefetch_b8
      Frag16 b;
      b.q[0] = *(const uint4*)(wRow);
      b.q[1] = *(const uint4*)(wRow + 16);
      acc[nt] = __builtin_amdgcn_wmma_f32_16x16x32_bf16(
          false, a.v, false, b.v, (short)0, acc[nt], false, false);
    }

    asm volatile("s_wait_asynccnt 0x0" ::: "memory");
    __syncthreads();
    buf ^= 1;
  }

  #pragma unroll
  for (int nt = 0; nt < 4; ++nt) {
    const int col = cBase + nt * 16 + lrow;
    const float bv = bias[col];
    #pragma unroll
    for (int i = 0; i < 8; ++i) {
      const int r = rBase0 + mHalf + (hi ? 8 + i : i);
      out[(size_t)r * NC + col] = acc[nt][i] + bv;
    }
  }
}

// ---------------------------------------------------------------------------
// Persistent bidirectional-capable GRU chain.  One workgroup (8 waves) runs
// all T steps.  h kept in LDS (fp32 + bf16 mirror, padded row stride).
// Each wave owns HDIM/8 hidden columns and computes its own r/z/n gate
// columns, so only two barriers per step.
//   xg : (16*T, 3*HDIM) fp32  (precomputed x@Wih^T + bih)
//   Whh: (3*HDIM, HDIM) bf16
//   bhh: (3*HDIM) fp32
//   out: (16*T, outStride) fp32, written at [.. , outOff + col]
// ---------------------------------------------------------------------------
template<int HDIM>
__global__ __launch_bounds__(256)
void gru_seq(const float* __restrict__ xg,
             const unsigned short* __restrict__ Whh,
             const float* __restrict__ bhh,
             float* __restrict__ out,
             int T, int reverse, int outStride, int outOff) {
  constexpr int CPW = HDIM / 8;       // columns per wave
  constexpr int NT  = CPW / 16;       // 16-wide tiles per gate per wave
  constexpr int HP  = HDIM + 16;      // padded LDS row stride (bank rotate)

  __shared__ float          hf[16 * HP];
  __shared__ unsigned short hb[16 * HP];

  const int wave  = threadIdx.x >> 5;
  const int lane  = threadIdx.x & 31;
  const int lrow  = lane & 15;
  const bool hi   = lane >= 16;
  const int waveCol = wave * CPW;
  const int khalfA = hi ? 8 : 0;
  const int khalfB = hi ? 8 : 0;
  const v8f vzero = {0.f,0.f,0.f,0.f,0.f,0.f,0.f,0.f};

  for (int i = threadIdx.x; i < 16 * HP; i += 256) { hf[i] = 0.f; hb[i] = 0; }
  __syncthreads();

  for (int t = 0; t < T; ++t) {
    const int tt = reverse ? (T - 1 - t) : t;

    v8f acc[3][NT];
    #pragma unroll
    for (int g = 0; g < 3; ++g)
      #pragma unroll
      for (int n = 0; n < NT; ++n) acc[g][n] = vzero;

    for (int k0 = 0; k0 < HDIM; k0 += 32) {
      Frag16 a;                                  // A = h (16 x HDIM) from LDS
      const unsigned short* ar = hb + lrow * HP + k0 + khalfA;
      a.q[0] = *(const uint4*)(ar);
      a.q[1] = *(const uint4*)(ar + 16);
      #pragma unroll
      for (int g = 0; g < 3; ++g) {
        #pragma unroll
        for (int n = 0; n < NT; ++n) {
          const int colW = g * HDIM + waveCol + n * 16 + lrow;
          const unsigned short* wr =
              Whh + (size_t)colW * HDIM + k0 + khalfB;
          Frag16 b;
          b.q[0] = *(const uint4*)(wr);
          b.q[1] = *(const uint4*)(wr + 16);
          acc[g][n] = __builtin_amdgcn_wmma_f32_16x16x32_bf16(
              false, a.v, false, b.v, (short)0, acc[g][n], false, false);
        }
      }
    }

    // gates (fp32); reads of hf/hb complete before the barrier
    float hnew[NT][8];
    #pragma unroll
    for (int n = 0; n < NT; ++n) {
      const int col = waveCol + n * 16 + lrow;
      #pragma unroll
      for (int i = 0; i < 8; ++i) {
        const int bb = hi ? 8 + i : i;                 // batch row (M index)
        const size_t rbase = ((size_t)bb * T + tt) * (size_t)(3 * HDIM);
        const float xr = xg[rbase + col];
        const float xz = xg[rbase + HDIM + col];
        const float xn = xg[rbase + 2 * HDIM + col];
        const float gr = acc[0][n][i] + bhh[col];
        const float gz = acc[1][n][i] + bhh[HDIM + col];
        const float gn = acc[2][n][i] + bhh[2 * HDIM + col];
        const float r  = sigmoidf_(xr + gr);
        const float z  = sigmoidf_(xz + gz);
        const float nn = tanhf(xn + r * gn);
        const float ho = hf[bb * HP + col];
        hnew[n][i] = (1.f - z) * nn + z * ho;
      }
    }
    __syncthreads();
    #pragma unroll
    for (int n = 0; n < NT; ++n) {
      const int col = waveCol + n * 16 + lrow;
      #pragma unroll
      for (int i = 0; i < 8; ++i) {
        const int bb = hi ? 8 + i : i;
        const float v = hnew[n][i];
        hf[bb * HP + col] = v;
        hb[bb * HP + col] = f2bf(v);
        out[((size_t)bb * T + tt) * outStride + outOff + col] = v;
      }
    }
    __syncthreads();
  }
}

// ---------------------------------------------------------------------------
// BiDAF attention #1.  Block per (t,n), 64 threads (J=64).
//   S[j] = C[n,t]·w1 + Q[n,j]·w2 + (C[n,t]*w3)·Q[n,j] + b
//   c2q[n,t,:] = softmax_j(S) @ Q[n];  mrow[n,t] = max_j S
// ---------------------------------------------------------------------------
__global__ __launch_bounds__(64)
void attn1_kernel(const float* __restrict__ Cc, const float* __restrict__ Qq,
                  const float* __restrict__ w, const float* __restrict__ bp,
                  float* __restrict__ c2q, float* __restrict__ mrow) {
  const int n = blockIdx.y, t = blockIdx.x, j = threadIdx.x;
  __shared__ float crow[512], cw3[512], sP[64], red[64];
  const float* c = Cc + ((size_t)n * 512 + t) * 512;
  for (int d = j; d < 512; d += 64) {
    const float cv = c[d];
    crow[d] = cv;
    cw3[d]  = cv * w[1024 + d];
  }
  __syncthreads();
  const float* q = Qq + ((size_t)n * 64 + j) * 512;
  float am = 0.f, aq = 0.f, ac = 0.f;
  for (int d = 0; d < 512; ++d) {
    const float qv = q[d];
    am += cw3[d] * qv;
    aq += w[512 + d] * qv;
    ac += crow[d] * w[d];
  }
  const float S = ac + aq + am + bp[0];
  red[j] = S; __syncthreads();
  for (int s = 32; s > 0; s >>= 1) {
    if (j < s) red[j] = fmaxf(red[j], red[j + s]);
    __syncthreads();
  }
  const float mx = red[0]; __syncthreads();
  const float e = __expf(S - mx);
  sP[j] = e; red[j] = e; __syncthreads();
  for (int s = 32; s > 0; s >>= 1) {
    if (j < s) red[j] += red[j + s];
    __syncthreads();
  }
  const float inv = 1.f / red[0];
  if (j == 0) mrow[n * 512 + t] = mx;
  __syncthreads();
  sP[j] *= inv; __syncthreads();
  for (int d0 = 0; d0 < 512; d0 += 64) {
    const int d = d0 + j;
    float s = 0.f;
    for (int jj = 0; jj < 64; ++jj)
      s += sP[jj] * Qq[((size_t)n * 64 + jj) * 512 + d];
    c2q[((size_t)n * 512 + t) * 512 + d] = s;
  }
}

// ---------------------------------------------------------------------------
// outvec[n,d] = sum_t softmax_t(mrow[n,:])[t] * X[n,t,d]    (T=512, D=512)
// ---------------------------------------------------------------------------
__global__ __launch_bounds__(256)
void softmax_wsum(const float* __restrict__ mrow, const float* __restrict__ X,
                  float* __restrict__ outvec, int T, int D) {
  const int n = blockIdx.x, tid = threadIdx.x;
  __shared__ float a[512];
  __shared__ float red[256];
  float m = -1e30f;
  for (int t = tid; t < T; t += 256) m = fmaxf(m, mrow[n * T + t]);
  red[tid] = m; __syncthreads();
  for (int s = 128; s > 0; s >>= 1) {
    if (tid < s) red[tid] = fmaxf(red[tid], red[tid + s]);
    __syncthreads();
  }
  const float mx = red[0]; __syncthreads();
  float sum = 0.f;
  for (int t = tid; t < T; t += 256) {
    const float e = __expf(mrow[n * T + t] - mx);
    a[t] = e; sum += e;
  }
  red[tid] = sum; __syncthreads();
  for (int s = 128; s > 0; s >>= 1) {
    if (tid < s) red[tid] += red[tid + s];
    __syncthreads();
  }
  const float inv = 1.f / red[0]; __syncthreads();
  for (int t = tid; t < T; t += 256) a[t] *= inv;
  __syncthreads();
  for (int d = tid; d < D; d += 256) {
    float s = 0.f;
    for (int t = 0; t < T; ++t)
      s += a[t] * X[((size_t)n * T + t) * D + d];
    outvec[n * D + d] = s;
  }
}

// ---------------------------------------------------------------------------
// G(,+=) relu(concat([X, c2q, X*c2q, q2c*c2q]))  + bf16 mirror for GEMM A.
// ---------------------------------------------------------------------------
__global__ __launch_bounds__(256)
void build_g(const float* __restrict__ X, const float* __restrict__ c2q,
             const float* __restrict__ q2c, float* __restrict__ G,
             unsigned short* __restrict__ Gbf, int addMode) {
  const int n = blockIdx.y, t = blockIdx.x, tid = threadIdx.x;
  const size_t r = (size_t)n * 512 + t;
  float* gr = G + r * 2048;
  unsigned short* gb = Gbf + r * 2048;
  for (int d = tid; d < 512; d += 256) {
    const float x  = X[r * 512 + d];
    const float cq = c2q[r * 512 + d];
    const float qc = q2c[n * 512 + d];
    float g0 = fmaxf(x, 0.f);
    float g1 = fmaxf(cq, 0.f);
    float g2 = fmaxf(x * cq, 0.f);
    float g3 = fmaxf(qc * cq, 0.f);
    if (addMode) {
      g0 += gr[d]; g1 += gr[512 + d]; g2 += gr[1024 + d]; g3 += gr[1536 + d];
    }
    gr[d] = g0; gr[512 + d] = g1; gr[1024 + d] = g2; gr[1536 + d] = g3;
    gb[d] = f2bf(g0); gb[512 + d] = f2bf(g1);
    gb[1024 + d] = f2bf(g2); gb[1536 + d] = f2bf(g3);
  }
}

// ---------------------------------------------------------------------------
// Self-attention over ec (S2: T x T per (n,t) row).  Block per (t,n),
// 256 threads, 2 s-columns per thread.
// ---------------------------------------------------------------------------
__global__ __launch_bounds__(256)
void attn2_kernel(const float* __restrict__ ec, const float* __restrict__ cw,
                  const float* __restrict__ qw, const float* __restrict__ mw,
                  const float* __restrict__ cb, const float* __restrict__ qb,
                  const float* __restrict__ mb,
                  float* __restrict__ c2q2, float* __restrict__ m2) {
  const int n = blockIdx.y, t = blockIdx.x, tid = threadIdx.x;
  __shared__ float em[512];
  __shared__ float sP[512];
  __shared__ float red[256];
  const float* et = ec + ((size_t)n * 512 + t) * 512;
  float part = 0.f;
  for (int d = tid; d < 512; d += 256) {
    const float v = et[d];
    em[d] = v * mw[d];
    part += v * cw[d];
  }
  red[tid] = part; __syncthreads();
  for (int s = 128; s > 0; s >>= 1) {
    if (tid < s) red[tid] += red[tid + s];
    __syncthreads();
  }
  const float cdot = red[0] + cb[0] + qb[0] + mb[0];
  __syncthreads();
  float Sv[2];
  #pragma unroll
  for (int ss = 0; ss < 2; ++ss) {
    const int s = tid + ss * 256;
    const float* es = ec + ((size_t)n * 512 + s) * 512;
    float am = 0.f, aq = 0.f;
    for (int d = 0; d < 512; ++d) {
      const float v = es[d];
      am += em[d] * v;
      aq += qw[d] * v;
    }
    Sv[ss] = cdot + aq + am;
  }
  float mx = fmaxf(Sv[0], Sv[1]);
  red[tid] = mx; __syncthreads();
  for (int s = 128; s > 0; s >>= 1) {
    if (tid < s) red[tid] = fmaxf(red[tid], red[tid + s]);
    __syncthreads();
  }
  mx = red[0]; __syncthreads();
  const float e0 = __expf(Sv[0] - mx), e1 = __expf(Sv[1] - mx);
  sP[tid] = e0; sP[tid + 256] = e1;
  red[tid] = e0 + e1; __syncthreads();
  for (int s = 128; s > 0; s >>= 1) {
    if (tid < s) red[tid] += red[tid + s];
    __syncthreads();
  }
  const float inv = 1.f / red[0];
  if (tid == 0) m2[n * 512 + t] = mx;
  __syncthreads();
  sP[tid] *= inv; sP[tid + 256] *= inv;
  __syncthreads();
  for (int d0 = 0; d0 < 512; d0 += 256) {
    const int d = d0 + tid;
    float s = 0.f;
    for (int ss = 0; ss < 512; ++ss)
      s += sP[ss] * ec[((size_t)n * 512 + ss) * 512 + d];
    c2q2[((size_t)n * 512 + t) * 512 + d] = s;
  }
}

// ---------------------------------------------------------------------------
// GMbf = bf16(concat([G, M], -1))   (rows 8192, widths 2048 + 1024)
// ---------------------------------------------------------------------------
__global__ __launch_bounds__(256)
void build_gm(const float* __restrict__ G, const float* __restrict__ M,
              unsigned short* __restrict__ GMbf, size_t total) {
  const size_t i = (size_t)blockIdx.x * blockDim.x + threadIdx.x;
  if (i >= total) return;
  const size_t r = i / 3072;
  const size_t d = i % 3072;
  const float v = (d < 2048) ? G[r * 2048 + d] : M[r * 1024 + (d - 2048)];
  GMbf[i] = f2bf(v);
}

// ---------------------------------------------------------------------------
// out[r] = X[r,:]·w + b    (row dot, block reduce)
// ---------------------------------------------------------------------------
__global__ __launch_bounds__(256)
void proj_kernel(const float* __restrict__ X, const float* __restrict__ w,
                 const float* __restrict__ b, float* __restrict__ out, int D) {
  const int r = blockIdx.x, tid = threadIdx.x;
  __shared__ float red[256];
  float s = 0.f;
  for (int d = tid; d < D; d += 256) s += X[(size_t)r * D + d] * w[d];
  red[tid] = s; __syncthreads();
  for (int st = 128; st > 0; st >>= 1) {
    if (tid < st) red[tid] += red[tid + st];
    __syncthreads();
  }
  if (tid == 0) out[r] = red[0] + b[0];
}

// ---------------------------------------------------------------------------
// Host-side orchestration
// ---------------------------------------------------------------------------
extern "C" void kernel_launch(void* const* d_in, const int* in_sizes, int n_in,
                              void* d_out, int out_size, void* d_ws,
                              size_t ws_size, hipStream_t stream) {
  (void)in_sizes; (void)n_in; (void)out_size; (void)ws_size;
  constexpr int N = 16, T = 512, Dm = 512;     // Dm = 2H
  constexpr size_t R = (size_t)N * T;          // 8192 flattened rows

  const float* Cc      = (const float*)d_in[0];
  const float* Qq      = (const float*)d_in[1];
  const float* embW_w  = (const float*)d_in[2];
  const float* embW_b  = (const float*)d_in[3];
  const float* cW_w    = (const float*)d_in[4];
  const float* cW_b    = (const float*)d_in[5];
  const float* qW_w    = (const float*)d_in[6];
  const float* qW_b    = (const float*)d_in[7];
  const float* mW_w    = (const float*)d_in[8];
  const float* mW_b    = (const float*)d_in[9];
  const float* self_Wih = (const float*)d_in[10];
  const float* self_Whh = (const float*)d_in[11];
  const float* self_bih = (const float*)d_in[12];
  const float* self_bhh = (const float*)d_in[13];
  const float* mod_Wih  = (const float*)d_in[14];
  const float* mod_Whh  = (const float*)d_in[15];
  const float* mod_bih  = (const float*)d_in[16];
  const float* mod_bhh  = (const float*)d_in[17];
  const float* p2g_Wih  = (const float*)d_in[18];
  const float* p2g_Whh  = (const float*)d_in[19];
  const float* p2g_bih  = (const float*)d_in[20];
  const float* p2g_bhh  = (const float*)d_in[21];
  const float* p1_w     = (const float*)d_in[22];
  const float* p1_b     = (const float*)d_in[23];
  const float* p2_w     = (const float*)d_in[24];
  const float* p2_b     = (const float*)d_in[25];
  float* outp = (float*)d_out;

  // workspace carve-up (256B aligned)
  char* base = (char*)d_ws;
  size_t off = 0;
  auto alloc = [&](size_t bytes) -> char* {
    char* p = base + off;
    off = (off + bytes + 255) & ~(size_t)255;
    return p;
  };
  float* G            = (float*)alloc(R * 2048 * 4);
  unsigned short* Gbf = (unsigned short*)alloc(R * 2048 * 2);
  float* c2q          = (float*)alloc(R * 512 * 4);     // reused for c2q2
  float* q2c          = (float*)alloc((size_t)N * 512 * 4);
  float* q2c2         = (float*)alloc((size_t)N * 512 * 4);
  float* mrow         = (float*)alloc((size_t)N * T * 4);
  float* m2           = (float*)alloc((size_t)N * T * 4);
  float* ec           = (float*)alloc(R * 512 * 4);
  float* xg           = (float*)alloc(R * 1536 * 4);    // max 3*HDIM = 1536
  float* Mb           = (float*)alloc(R * 1024 * 4);
  float* M2b          = (float*)alloc(R * 1024 * 4);
  unsigned short* GMbf     = (unsigned short*)alloc(R * 3072 * 2);
  unsigned short* wSelfIh  = (unsigned short*)alloc((size_t)2 * 768 * 2048 * 2);
  unsigned short* wSelfHh  = (unsigned short*)alloc((size_t)2 * 768 * 256 * 2);
  unsigned short* wModIh   = (unsigned short*)alloc((size_t)2 * 1536 * 2048 * 2);
  unsigned short* wModHh   = (unsigned short*)alloc((size_t)2 * 1536 * 512 * 2);
  unsigned short* wP2gIh   = (unsigned short*)alloc((size_t)2 * 1536 * 3072 * 2);
  unsigned short* wP2gHh   = (unsigned short*)alloc((size_t)2 * 1536 * 512 * 2);

  auto cvt = [&](const float* src, unsigned short* dst, size_t n) {
    cvt_f32_bf16<<<(unsigned)((n + 255) / 256), 256, 0, stream>>>(src, dst, n);
  };
  // weight conversions (L2-resident bf16 copies)
  cvt(self_Wih, wSelfIh, (size_t)2 * 768 * 2048);
  cvt(self_Whh, wSelfHh, (size_t)2 * 768 * 256);
  cvt(mod_Wih,  wModIh,  (size_t)2 * 1536 * 2048);
  cvt(mod_Whh,  wModHh,  (size_t)2 * 1536 * 512);
  cvt(p2g_Wih,  wP2gIh,  (size_t)2 * 1536 * 3072);
  cvt(p2g_Whh,  wP2gHh,  (size_t)2 * 1536 * 512);

  // ---- stage 1: BiDAF attention -> G ----
  attn1_kernel<<<dim3(T, N), 64, 0, stream>>>(Cc, Qq, embW_w, embW_b, c2q, mrow);
  softmax_wsum<<<N, 256, 0, stream>>>(mrow, Cc, q2c, T, Dm);
  build_g<<<dim3(T, N), 256, 0, stream>>>(Cc, c2q, q2c, G, Gbf, 0);

  // ---- stage 2: self BiGRU (input 2048 -> hidden 256/dir) -> ec ----
  for (int dir = 0; dir < 2; ++dir) {
    gemm_bf16_nt<<<dim3(R / 32, 768 / 256), 256, 0, stream>>>(
        Gbf, wSelfIh + (size_t)dir * 768 * 2048, self_bih + dir * 768,
        xg, 2048, 768);
    gru_seq<256><<<1, 256, 0, stream>>>(
        xg, wSelfHh + (size_t)dir * 768 * 256, self_bhh + dir * 768,
        ec, T, dir, 512, dir * 256);
  }

  // ---- stage 3: self-attention over ec, G += self_G ----
  attn2_kernel<<<dim3(T, N), 256, 0, stream>>>(
      ec, cW_w, qW_w, mW_w, cW_b, qW_b, mW_b, c2q, m2);
  softmax_wsum<<<N, 256, 0, stream>>>(m2, ec, q2c2, T, Dm);
  build_g<<<dim3(T, N), 256, 0, stream>>>(ec, c2q, q2c2, G, Gbf, 1);

  // ---- stage 4: mod BiGRU (2048 -> 512/dir) -> M, p1 ----
  for (int dir = 0; dir < 2; ++dir) {
    gemm_bf16_nt<<<dim3(R / 32, 1536 / 256), 256, 0, stream>>>(
        Gbf, wModIh + (size_t)dir * 1536 * 2048, mod_bih + dir * 1536,
        xg, 2048, 1536);
    gru_seq<512><<<1, 256, 0, stream>>>(
        xg, wModHh + (size_t)dir * 1536 * 512, mod_bhh + dir * 1536,
        Mb, T, dir, 1024, dir * 512);
  }
  proj_kernel<<<(unsigned)R, 256, 0, stream>>>(Mb, p1_w, p1_b, outp, 1024);

  // ---- stage 5: p2g BiGRU on concat([G, M]) (3072 -> 512/dir) -> M2, p2 ----
  {
    const size_t total = R * 3072;
    build_gm<<<(unsigned)((total + 255) / 256), 256, 0, stream>>>(G, Mb, GMbf,
                                                                 total);
  }
  for (int dir = 0; dir < 2; ++dir) {
    gemm_bf16_nt<<<dim3(R / 32, 1536 / 256), 256, 0, stream>>>(
        GMbf, wP2gIh + (size_t)dir * 1536 * 3072, p2g_bih + dir * 1536,
        xg, 3072, 1536);
    gru_seq<512><<<1, 256, 0, stream>>>(
        xg, wP2gHh + (size_t)dir * 1536 * 512, p2g_bhh + dir * 1536,
        M2b, T, dir, 1024, dir * 512);
  }
  proj_kernel<<<(unsigned)R, 256, 0, stream>>>(M2b, p2_w, p2_b, outp + R, 1024);
}
